// POINTPLUS_77455440216466
// MI455X (gfx1250) — matrix operand order repaired and checked by reading, not compiled, not run on hardware
//
#include <hip/hip_runtime.h>

#define BATCH 4
#define NPTS  16384

typedef __attribute__((ext_vector_type(16))) __bf16        v16bf;
typedef __attribute__((ext_vector_type(8)))  float         v8f;
typedef __attribute__((ext_vector_type(4)))  unsigned int  u32x4;

union Frag { v16bf v; u32x4 q[2]; };

__device__ __forceinline__ unsigned short f2bf(float x) {
    unsigned int u = __float_as_uint(x);
    u += 0x7FFFu + ((u >> 16) & 1u);   // round-to-nearest-even
    return (unsigned short)(u >> 16);
}
__device__ __forceinline__ float bf2f(unsigned short u) {
    return __uint_as_float((unsigned int)u << 16);
}

#define GF_BN     1
#define GF_RELU   2
#define GF_MAXRED 4
#define GF_F32OUT 8

// ---------------------------------------------------------------------------
// Weight pre-pack: W (Cout,Cin) f32 -> bf16 fragments in WMMA B layout.
// Fragment index = ((ntile*nk + ks)*32 + lane), 16 bf16 (32B) per fragment.
// B 32x16 layout: col = lane&15; lanes 0-15 hold K 0..7,16..23 (2/vgpr), 16-31 hold 8..15,24..31.
// Cout padded to 64 (zero cols), K padded to 32 (zero rows).
// ---------------------------------------------------------------------------
__global__ void k_packW(const float* __restrict__ W, int Cin, int Cout,
                        int nt, int nk, unsigned short* __restrict__ out)
{
    int t = blockIdx.x * blockDim.x + threadIdx.x;
    if (t >= nt * nk * 32) return;
    int lane  = t & 31;
    int rest  = t >> 5;
    int ks    = rest % nk;
    int ntile = rest / nk;
    int ncol  = ntile * 16 + (lane & 15);
    int half  = lane >> 4;
    unsigned short* o = out + (size_t)t * 16;
    #pragma unroll
    for (int i = 0; i < 8; ++i) {
        int kb = ks * 32 + (i >> 2) * 16 + half * 8 + (i & 3) * 2;
        float b0 = (ncol < Cout && kb     < Cin) ? W[(size_t)ncol * Cin + kb]     : 0.f;
        float b1 = (ncol < Cout && kb + 1 < Cin) ? W[(size_t)ncol * Cin + kb + 1] : 0.f;
        o[2 * i]     = f2bf(b0);
        o[2 * i + 1] = f2bf(b1);
    }
}

// Fused bias + BN + ReLU epilogue for one 16x16 accumulator tile.
__device__ __forceinline__ void gemm_epilogue(
    v8f acc, int ncol, int m0, int half,
    const float* __restrict__ bias,
    const float* __restrict__ bng, const float* __restrict__ bnb,
    const float* __restrict__ bnm, const float* __restrict__ bnv,
    void* __restrict__ Yv, int ldc, int Cout, int flags, int rows_per_batch)
{
    bool ok  = (ncol < Cout);
    float add = ok ? bias[ncol] : 0.f;
    float sc = 1.f, sh = 0.f;
    if ((flags & GF_BN) && ok) {
        float inv = rsqrtf(bnv[ncol] + 1e-5f);
        sc = bng[ncol] * inv;
        sh = bnb[ncol] - bnm[ncol] * sc;
    }
    #pragma unroll
    for (int r = 0; r < 8; ++r) {
        int mm = m0 + r + half * 8;
        float val = acc[r] + add;
        if (flags & GF_BN)   val = val * sc + sh;
        if (flags & GF_RELU) val = fmaxf(val, 0.f);
        if (flags & GF_MAXRED) {
            if (ok) {
                int bidx = mm / rows_per_batch;
                atomicMax((unsigned int*)((float*)Yv + (size_t)bidx * ldc + ncol),
                          __float_as_uint(val));
            }
        } else if (flags & GF_F32OUT) {
            if (ok) ((float*)Yv)[(size_t)mm * ldc + ncol] = val;
        } else {
            if (ok) ((unsigned short*)Yv)[(size_t)mm * ldc + ncol] = f2bf(val);
        }
    }
}

// ---------------------------------------------------------------------------
// Y = [relu][bn](X @ W^T + bias). One wave computes a 16x64 output tile:
// one A fragment (2x b128) reused across 4 B fragments (8x b128) -> 4 WMMA/K-step.
// X: bf16, row stride lda (multiple of 32, zero-padded K).
// NOTE: named pointer scalars (not arrays) so addrspace inference keeps
// all loads as global_load_b128 (flat_load would also tick DScnt).
// ---------------------------------------------------------------------------
__global__ void k_gemm(const unsigned short* __restrict__ X,
                       const unsigned short* __restrict__ Wp,
                       const float* __restrict__ bias,
                       const float* __restrict__ bng, const float* __restrict__ bnb,
                       const float* __restrict__ bnm, const float* __restrict__ bnv,
                       void* __restrict__ Yv, int lda, int nk, int ldc, int Cout,
                       int flags, int rows_per_batch)
{
    const int lane = threadIdx.x;      // one wave
    const int r16  = lane & 15;
    const int half = lane >> 4;
    const int m0   = blockIdx.x * 16;

    const unsigned short* xrow = X + (size_t)(m0 + r16) * lda;
    const size_t wstride = (size_t)nk * 512;   // elements per n-tile
    const unsigned short* wp0 = Wp + (size_t)(blockIdx.y * 4) * wstride + (size_t)lane * 16;
    const unsigned short* wp1 = wp0 + wstride;
    const unsigned short* wp2 = wp1 + wstride;
    const unsigned short* wp3 = wp2 + wstride;

    v8f acc0 = {}, acc1 = {}, acc2 = {}, acc3 = {};
    #pragma unroll 2
    for (int ks = 0; ks < nk; ++ks) {
        Frag a, b0, b1, b2, b3;
        const unsigned short* xk = xrow + ks * 32;
        a.q[0]  = *(const u32x4*)(xk + half * 8);        // K = k0+half*8 .. +7
        a.q[1]  = *(const u32x4*)(xk + 16 + half * 8);   // K = k0+16+half*8 .. +7
        b0.q[0] = ((const u32x4*)wp0)[0];
        b0.q[1] = ((const u32x4*)wp0)[1];
        b1.q[0] = ((const u32x4*)wp1)[0];
        b1.q[1] = ((const u32x4*)wp1)[1];
        b2.q[0] = ((const u32x4*)wp2)[0];
        b2.q[1] = ((const u32x4*)wp2)[1];
        b3.q[0] = ((const u32x4*)wp3)[0];
        b3.q[1] = ((const u32x4*)wp3)[1];
        wp0 += 512; wp1 += 512; wp2 += 512; wp3 += 512;
        acc0 = __builtin_amdgcn_wmma_f32_16x16x32_bf16(false, a.v, false, b0.v,
                                                       (short)0, acc0, false, false);
        acc1 = __builtin_amdgcn_wmma_f32_16x16x32_bf16(false, a.v, false, b1.v,
                                                       (short)0, acc1, false, false);
        acc2 = __builtin_amdgcn_wmma_f32_16x16x32_bf16(false, a.v, false, b2.v,
                                                       (short)0, acc2, false, false);
        acc3 = __builtin_amdgcn_wmma_f32_16x16x32_bf16(false, a.v, false, b3.v,
                                                       (short)0, acc3, false, false);
    }

    int nbase = blockIdx.y * 64 + r16;
    gemm_epilogue(acc0, nbase,      m0, half, bias, bng, bnb, bnm, bnv, Yv, ldc, Cout, flags, rows_per_batch);
    gemm_epilogue(acc1, nbase + 16, m0, half, bias, bng, bnb, bnm, bnv, Yv, ldc, Cout, flags, rows_per_batch);
    gemm_epilogue(acc2, nbase + 32, m0, half, bias, bng, bnb, bnm, bnv, Yv, ldc, Cout, flags, rows_per_batch);
    gemm_epilogue(acc3, nbase + 48, m0, half, bias, bng, bnb, bnm, bnv, Yv, ldc, Cout, flags, rows_per_batch);
}

// ---------------------------------------------------------------------------
// (B,3,N) -> pts3 (B,N,3) f32  +  ptsA (B*N, 32) bf16 zero-padded GEMM input
// ---------------------------------------------------------------------------
__global__ void k_prep_pts(const float* __restrict__ xyz, float* __restrict__ pts3,
                           unsigned short* __restrict__ ptsA)
{
    int t = blockIdx.x * blockDim.x + threadIdx.x;
    if (t >= BATCH * NPTS) return;
    int b = t / NPTS, n = t % NPTS;
    float x = xyz[((size_t)b * 3 + 0) * NPTS + n];
    float y = xyz[((size_t)b * 3 + 1) * NPTS + n];
    float z = xyz[((size_t)b * 3 + 2) * NPTS + n];
    pts3[(size_t)t * 3 + 0] = x;
    pts3[(size_t)t * 3 + 1] = y;
    pts3[(size_t)t * 3 + 2] = z;
    unsigned short* o = ptsA + (size_t)t * 32;
    o[0] = f2bf(x); o[1] = f2bf(y); o[2] = f2bf(z);
    #pragma unroll
    for (int c = 3; c < 32; ++c) o[c] = 0;
}

__global__ void k_make_trans(const float* __restrict__ h, float* __restrict__ trans,
                             float* __restrict__ out_trans)
{
    int t = threadIdx.x;
    if (t >= BATCH * 9) return;
    int b = t / 9, e = t % 9;
    float v = h[(size_t)b * 32 + e] + ((e == 0 || e == 4 || e == 8) ? 1.f : 0.f);
    trans[t] = v;
    out_trans[t] = v;
}

// l0_points = pts @ trans, stored bf16 (feature input of SA1)
__global__ void k_apply_trans(const float* __restrict__ pts3, const float* __restrict__ trans,
                              unsigned short* __restrict__ out)
{
    int t = blockIdx.x * blockDim.x + threadIdx.x;
    if (t >= BATCH * NPTS) return;
    int b = t / NPTS;
    float x = pts3[(size_t)t * 3 + 0];
    float y = pts3[(size_t)t * 3 + 1];
    float z = pts3[(size_t)t * 3 + 2];
    const float* T = trans + b * 9;
    out[(size_t)t * 3 + 0] = f2bf(x * T[0] + y * T[3] + z * T[6]);
    out[(size_t)t * 3 + 1] = f2bf(x * T[1] + y * T[4] + z * T[7]);
    out[(size_t)t * 3 + 2] = f2bf(x * T[2] + y * T[5] + z * T[8]);
}

__global__ void k_cvt_bf16(const float* __restrict__ x, unsigned short* __restrict__ y, int n)
{
    int t = blockIdx.x * blockDim.x + threadIdx.x;
    if (t < n) y[t] = f2bf(x[t]);
}

// ---------------------------------------------------------------------------
// Farthest point sampling: one workgroup per batch, LDS argmax per round.
// ---------------------------------------------------------------------------
__global__ void k_fps(const float* __restrict__ pts, int N, int npoint,
                      int* __restrict__ idx, float* __restrict__ dist)
{
    __shared__ float sv[256];
    __shared__ int   si[256];
    __shared__ int   s_far;
    int b = blockIdx.x;
    int tid = threadIdx.x;
    const float* P = pts + (size_t)b * N * 3;
    float* D = dist + (size_t)b * N;
    for (int i = tid; i < N; i += blockDim.x) D[i] = 1e10f;
    if (tid == 0) s_far = 0;
    __syncthreads();
    for (int j = 0; j < npoint; ++j) {
        int far = s_far;
        if (tid == 0) idx[b * npoint + j] = far;
        float cx = P[far * 3], cy = P[far * 3 + 1], cz = P[far * 3 + 2];
        float best = -1.f; int bi = 0;
        for (int i = tid; i < N; i += blockDim.x) {
            float dx = P[i * 3] - cx, dy = P[i * 3 + 1] - cy, dz = P[i * 3 + 2] - cz;
            float d = dx * dx + dy * dy + dz * dz;
            float dd = fminf(D[i], d);
            D[i] = dd;
            if (dd > best) { best = dd; bi = i; }
        }
        sv[tid] = best; si[tid] = bi;
        __syncthreads();
        for (int s = blockDim.x / 2; s > 0; s >>= 1) {
            if (tid < s && sv[tid + s] > sv[tid]) { sv[tid] = sv[tid + s]; si[tid] = si[tid + s]; }
            __syncthreads();
        }
        if (tid == 0) s_far = si[0];
        __syncthreads();
    }
}

__global__ void k_gather_xyz(const float* __restrict__ pts, const int* __restrict__ idx,
                             int N, int S, float* __restrict__ out)
{
    int t = blockIdx.x * blockDim.x + threadIdx.x;
    if (t >= BATCH * S) return;
    int b = t / S;
    int gi = idx[t];
    const float* p = pts + ((size_t)b * N + gi) * 3;
    out[(size_t)t * 3 + 0] = p[0];
    out[(size_t)t * 3 + 1] = p[1];
    out[(size_t)t * 3 + 2] = p[2];
}

// First nsample in-radius indices in scan order == reference sort+slice.
__global__ void k_ballquery(const float* __restrict__ pts, const float* __restrict__ nxyz,
                            int N, int S, int nsample, float r2, int* __restrict__ idx)
{
    int t = blockIdx.x * blockDim.x + threadIdx.x;
    if (t >= BATCH * S) return;
    int b = t / S;
    const float* P = pts + (size_t)b * N * 3;
    float qx = nxyz[(size_t)t * 3], qy = nxyz[(size_t)t * 3 + 1], qz = nxyz[(size_t)t * 3 + 2];
    int* o = idx + (size_t)t * nsample;
    int cnt = 0;
    for (int i = 0; i < N && cnt < nsample; ++i) {
        float dx = P[i * 3] - qx, dy = P[i * 3 + 1] - qy, dz = P[i * 3 + 2] - qz;
        float d = dx * dx + dy * dy + dz * dz;
        if (d <= r2) o[cnt++] = i;
    }
    if (cnt == 0) { o[0] = 0; cnt = 1; }
    int f = o[0];
    for (int j = cnt; j < nsample; ++j) o[j] = f;
}

// grouped[b,s,j,:] = concat(pts[idx]-new_xyz, feats[idx]), bf16, zero-padded row
__global__ void k_group(const float* __restrict__ pts3, const unsigned short* __restrict__ feats,
                        const float* __restrict__ nxyz, const int* __restrict__ idx,
                        int N, int S, int ns, int Cf, int stride,
                        unsigned short* __restrict__ out)
{
    int t = blockIdx.x * blockDim.x + threadIdx.x;
    if (t >= BATCH * S * ns) return;
    int bs = t / ns;
    int b  = bs / S;
    int gi = idx[t];
    const float* p = pts3 + ((size_t)b * N + gi) * 3;
    const float* q = nxyz + (size_t)bs * 3;
    unsigned short* o = out + (size_t)t * stride;
    o[0] = f2bf(p[0] - q[0]);
    o[1] = f2bf(p[1] - q[1]);
    o[2] = f2bf(p[2] - q[2]);
    const unsigned short* f = feats + ((size_t)b * N + gi) * Cf;
    for (int c = 0; c < Cf; ++c) o[3 + c] = f[c];
    for (int c = 3 + Cf; c < stride; ++c) o[c] = 0;
}

// Max over nsample; post-ReLU bf16 >= 0 so raw ushort compare == float compare.
__global__ void k_maxpool(const unsigned short* __restrict__ x, int ns, int C, int BS,
                          unsigned short* __restrict__ y)
{
    int t = blockIdx.x * blockDim.x + threadIdx.x;
    if (t >= BS * C) return;
    int bs = t / C, c = t % C;
    const unsigned short* p = x + (size_t)bs * ns * C + c;
    unsigned short m = p[0];
    for (int j = 1; j < ns; ++j) {
        unsigned short cand = p[(size_t)j * C];
        m = (cand > m) ? cand : m;
    }
    y[t] = m;
}

// 3-NN inverse-(sq)distance interpolation + optional skip-concat (bf16 features)
__global__ void k_threeinterp(const float* __restrict__ xyz1, const float* __restrict__ xyz2,
                              const unsigned short* __restrict__ p1,
                              const unsigned short* __restrict__ p2,
                              int S1, int S2, int C1, int C2, unsigned short* __restrict__ out)
{
    int t = blockIdx.x * blockDim.x + threadIdx.x;
    if (t >= BATCH * S1) return;
    int b = t / S1;
    float qx = xyz1[(size_t)t * 3], qy = xyz1[(size_t)t * 3 + 1], qz = xyz1[(size_t)t * 3 + 2];
    const float* X2 = xyz2 + (size_t)b * S2 * 3;
    float d0 = 1e30f, d1 = 1e30f, d2 = 1e30f;
    int   i0 = 0, i1 = 0, i2 = 0;
    for (int i = 0; i < S2; ++i) {
        float dx = X2[i * 3] - qx, dy = X2[i * 3 + 1] - qy, dz = X2[i * 3 + 2] - qz;
        float d = dx * dx + dy * dy + dz * dz;
        if (d < d0)      { d2 = d1; i2 = i1; d1 = d0; i1 = i0; d0 = d; i0 = i; }
        else if (d < d1) { d2 = d1; i2 = i1; d1 = d;  i1 = i; }
        else if (d < d2) { d2 = d;  i2 = i; }
    }
    float w0 = 1.f / (d0 + 1e-8f), w1 = 1.f / (d1 + 1e-8f), w2 = 1.f / (d2 + 1e-8f);
    float ws = w0 + w1 + w2;
    w0 /= ws; w1 /= ws; w2 /= ws;
    unsigned short* o = out + (size_t)t * (C1 + C2);
    if (p1) for (int c = 0; c < C1; ++c) o[c] = p1[(size_t)t * C1 + c];
    const unsigned short* f0 = p2 + ((size_t)b * S2 + i0) * C2;
    const unsigned short* f1 = p2 + ((size_t)b * S2 + i1) * C2;
    const unsigned short* f2 = p2 + ((size_t)b * S2 + i2) * C2;
    for (int c = 0; c < C2; ++c)
        o[C1 + c] = f2bf(w0 * bf2f(f0[c]) + w1 * bf2f(f1[c]) + w2 * bf2f(f2[c]));
}

__global__ void k_logsoftmax(const float* __restrict__ x, int ldx, int C, int M,
                             float* __restrict__ out)
{
    int t = blockIdx.x * blockDim.x + threadIdx.x;
    if (t >= M) return;
    const float* p = x + (size_t)t * ldx;
    float mx = p[0];
    for (int c = 1; c < C; ++c) mx = fmaxf(mx, p[c]);
    float s = 0.f;
    for (int c = 0; c < C; ++c) s += expf(p[c] - mx);
    float ls = logf(s);
    for (int c = 0; c < C; ++c) out[(size_t)t * C + c] = p[c] - mx - ls;
}

// ===========================================================================
// Host side
// ===========================================================================

// JAX pytree (sorted-key) flattening of the params dict.
enum {
    IN_XYZ = 0, IN_ALIGN = 1,
    FP1 = 2, FP2 = 20, FP3 = 32, FP4 = 44,
    HEAD1_W = 56, HEAD1_B = 57, HEAD2_W = 58, HEAD2_B = 59,
    HBN_B = 60, HBN_G = 61, HBN_M = 62, HBN_V = 63,
    SA1 = 64, SA2 = 82, SA3 = 100, SA4 = 118,
    STN_BN1 = 136, STN_BN2 = 140, STN_BN3 = 144, STN_BN4 = 148, STN_BN5 = 152,
    STN_C1 = 156, STN_C2 = 158, STN_C3 = 160, STN_F1 = 162, STN_F2 = 164, STN_F3 = 166,
    N_EXPECTED = 168
};

struct Lyr { const float *W, *b, *g, *bt, *m, *v; int co; };

static inline Lyr mk_layer(void* const* d, int base, int co)
{
    Lyr L;
    L.bt = (const float*)d[base + 0];
    L.g  = (const float*)d[base + 1];
    L.m  = (const float*)d[base + 2];
    L.v  = (const float*)d[base + 3];
    L.W  = (const float*)d[base + 4];
    L.b  = (const float*)d[base + 5];
    L.co = co;
    return L;
}

struct Pk { const unsigned short* wp; int nk; int coutPad; };

static Pk pack_w(hipStream_t s, const float* W, int Cin, int Cout, unsigned short*& cur)
{
    int kpad = (Cin + 31) & ~31;
    int cpad = (Cout + 63) & ~63;          // 4 n-tiles per wave
    int nt = cpad / 16, nk = kpad / 32;
    int total = nt * nk * 32;
    k_packW<<<(total + 255) / 256, 256, 0, s>>>(W, Cin, Cout, nt, nk, cur);
    Pk p{cur, nk, cpad};
    cur = (unsigned short*)((((uintptr_t)(cur + (size_t)total * 16)) + 255) & ~(uintptr_t)255);
    return p;
}

static inline void gemm(hipStream_t s, const unsigned short* X, int lda, const Pk& pk,
                        const float* bias, const float* g, const float* bt,
                        const float* m, const float* v,
                        void* Y, int ldc, int Mpad, int Cout, int flags, int rpb)
{
    dim3 gr((unsigned)(Mpad / 16), (unsigned)(pk.coutPad / 64));
    k_gemm<<<gr, 32, 0, s>>>(X, pk.wp, bias, g, bt, m, v, Y, lda, pk.nk, ldc, Cout, flags, rpb);
}

static void run_sa(hipStream_t s, void* const* d,
                   const float* xyz_in, int Nin, const unsigned short* feats, int Cf,
                   int S, float radius, int ns, int mlp_base, const int* outs, int nl,
                   const Pk* pks,
                   float* xyz_out, unsigned short* pooled,
                   int* fpsidx, int* ballidx, float* dist,
                   unsigned short* bufA, unsigned short* bufB)
{
    k_fps<<<BATCH, 256, 0, s>>>(xyz_in, Nin, S, fpsidx, dist);
    int tS = BATCH * S;
    k_gather_xyz<<<(tS + 255) / 256, 256, 0, s>>>(xyz_in, fpsidx, Nin, S, xyz_out);
    k_ballquery<<<(tS + 255) / 256, 256, 0, s>>>(xyz_in, xyz_out, Nin, S, ns,
                                                 radius * radius, ballidx);
    int rows = BATCH * S * ns;
    int lda  = (3 + Cf + 31) & ~31;
    k_group<<<(rows + 255) / 256, 256, 0, s>>>(xyz_in, feats, xyz_out, ballidx,
                                               Nin, S, ns, Cf, lda, bufA);
    const unsigned short* cur = bufA;
    for (int l = 0; l < nl; ++l) {
        Lyr L = mk_layer(d, mlp_base + 6 * l, outs[l]);
        unsigned short* out = (cur == bufA) ? bufB : bufA;
        gemm(s, cur, lda, pks[l], L.b, L.g, L.bt, L.m, L.v, out, L.co,
             rows, L.co, GF_BN | GF_RELU, 0);
        cur = out;
        lda = L.co;   // bf16 Cout is always a multiple of 32
    }
    int C = outs[nl - 1];
    k_maxpool<<<(BATCH * S * C + 255) / 256, 256, 0, s>>>(cur, ns, C, BATCH * S, pooled);
}

static void run_fp(hipStream_t s, void* const* d,
                   const float* xyz1, int S1, const float* xyz2, int S2,
                   const unsigned short* p1, int C1, const unsigned short* p2, int C2,
                   int mlp_base, const int* outs, int nl, const Pk* pks,
                   unsigned short* bufA, unsigned short* bufB, unsigned short* dst)
{
    int M = BATCH * S1;
    k_threeinterp<<<(M + 255) / 256, 256, 0, s>>>(xyz1, xyz2, p1, p2, S1, S2, C1, C2, bufA);
    const unsigned short* cur = bufA;
    int lda = C1 + C2;   // always a multiple of 32 here
    for (int l = 0; l < nl; ++l) {
        Lyr L = mk_layer(d, mlp_base + 6 * l, outs[l]);
        unsigned short* out = (l == nl - 1) ? dst : ((cur == bufA) ? bufB : bufA);
        gemm(s, cur, lda, pks[l], L.b, L.g, L.bt, L.m, L.v, out, L.co,
             M, L.co, GF_BN | GF_RELU, 0);
        cur = out;
        lda = L.co;
    }
}

extern "C" void kernel_launch(void* const* d_in, const int* in_sizes, int n_in,
                              void* d_out, int out_size, void* d_ws, size_t ws_size,
                              hipStream_t stream)
{
    (void)in_sizes; (void)out_size; (void)ws_size;
    if (n_in < N_EXPECTED) return;

    #define F(i) ((const float*)d_in[(i)])
    #define BN4(base) F((base) + 1), F((base) + 0), F((base) + 2), F((base) + 3)

    char* wptr = (char*)d_ws;
    auto alloc = [&](size_t nbytes) -> void* {
        void* p = (void*)wptr;
        wptr += (nbytes + 255) & ~(size_t)255;
        return p;
    };
    const int BN = BATCH * NPTS;

    float*          pts3    = (float*)alloc((size_t)BN * 3 * 4);
    unsigned short* ptsA    = (unsigned short*)alloc((size_t)BN * 32 * 2);
    unsigned short* l0p     = (unsigned short*)alloc((size_t)BN * 3 * 2);
    float*          hmax    = (float*)alloc((size_t)16 * 1024 * 4);
    unsigned short* hmaxbf  = (unsigned short*)alloc((size_t)16 * 1024 * 2);
    unsigned short* f1o     = (unsigned short*)alloc((size_t)16 * 512 * 2);
    unsigned short* f2o     = (unsigned short*)alloc((size_t)16 * 256 * 2);
    float*          f3o     = (float*)alloc((size_t)16 * 32 * 4);
    float*          trans   = (float*)alloc(64 * 4);
    float*          dist    = (float*)alloc((size_t)BN * 4);
    int*            fpsidx  = (int*)alloc((size_t)BATCH * 1024 * 4);
    int*            ballidx = (int*)alloc((size_t)BATCH * 1024 * 32 * 4);
    float*          l1_xyz  = (float*)alloc((size_t)BATCH * 1024 * 3 * 4);
    float*          l2_xyz  = (float*)alloc((size_t)BATCH * 256 * 3 * 4);
    float*          l3_xyz  = (float*)alloc((size_t)BATCH * 64 * 3 * 4);
    float*          l4_xyz  = (float*)alloc((size_t)BATCH * 16 * 3 * 4);
    unsigned short* l1_p    = (unsigned short*)alloc((size_t)BATCH * 1024 * 64 * 2);
    unsigned short* l2_p    = (unsigned short*)alloc((size_t)BATCH * 256 * 128 * 2);
    unsigned short* l3_p    = (unsigned short*)alloc((size_t)BATCH * 64 * 256 * 2);
    unsigned short* l4_p    = (unsigned short*)alloc((size_t)BATCH * 16 * 512 * 2);
    unsigned short* l3_p2   = (unsigned short*)alloc((size_t)BATCH * 64 * 256 * 2);
    unsigned short* l2_p2   = (unsigned short*)alloc((size_t)BATCH * 256 * 256 * 2);
    unsigned short* l1_p2   = (unsigned short*)alloc((size_t)BATCH * 1024 * 128 * 2);
    float*          logits  = (float*)alloc((size_t)BN * 32 * 4);
    unsigned short* BIGA    = (unsigned short*)alloc((size_t)8388608 * 2);
    unsigned short* BIGB    = (unsigned short*)alloc((size_t)8388608 * 2);
    unsigned short* wcur    = (unsigned short*)alloc(0);   // weight pack region tail

    // --- weight pre-pack (stream-ordered, runs before the GEMMs that use it) ---
    Pk pc1 = pack_w(stream, F(STN_C1), 3,    64,   wcur);
    Pk pc2 = pack_w(stream, F(STN_C2), 64,   128,  wcur);
    Pk pc3 = pack_w(stream, F(STN_C3), 128,  1024, wcur);
    Pk pf1 = pack_w(stream, F(STN_F1), 1024, 512,  wcur);
    Pk pf2 = pack_w(stream, F(STN_F2), 512,  256,  wcur);
    Pk pf3 = pack_w(stream, F(STN_F3), 256,  9,    wcur);
    Pk psa1[3] = { pack_w(stream, F(SA1 + 4),  6,   32,  wcur),
                   pack_w(stream, F(SA1 + 10), 32,  32,  wcur),
                   pack_w(stream, F(SA1 + 16), 32,  64,  wcur) };
    Pk psa2[3] = { pack_w(stream, F(SA2 + 4),  67,  64,  wcur),
                   pack_w(stream, F(SA2 + 10), 64,  64,  wcur),
                   pack_w(stream, F(SA2 + 16), 64,  128, wcur) };
    Pk psa3[3] = { pack_w(stream, F(SA3 + 4),  131, 128, wcur),
                   pack_w(stream, F(SA3 + 10), 128, 128, wcur),
                   pack_w(stream, F(SA3 + 16), 128, 256, wcur) };
    Pk psa4[3] = { pack_w(stream, F(SA4 + 4),  259, 256, wcur),
                   pack_w(stream, F(SA4 + 10), 256, 256, wcur),
                   pack_w(stream, F(SA4 + 16), 256, 512, wcur) };
    Pk pfp4[2] = { pack_w(stream, F(FP4 + 4),  768, 256, wcur),
                   pack_w(stream, F(FP4 + 10), 256, 256, wcur) };
    Pk pfp3[2] = { pack_w(stream, F(FP3 + 4),  384, 256, wcur),
                   pack_w(stream, F(FP3 + 10), 256, 256, wcur) };
    Pk pfp2[2] = { pack_w(stream, F(FP2 + 4),  320, 256, wcur),
                   pack_w(stream, F(FP2 + 10), 256, 128, wcur) };
    Pk pfp1[3] = { pack_w(stream, F(FP1 + 4),  128, 128, wcur),
                   pack_w(stream, F(FP1 + 10), 128, 128, wcur),
                   pack_w(stream, F(FP1 + 16), 128, 128, wcur) };
    Pk ph1 = pack_w(stream, F(HEAD1_W), 128, 128, wcur);
    Pk ph2 = pack_w(stream, F(HEAD2_W), 128, 7,   wcur);

    // --- STN ---
    k_prep_pts<<<(BN + 255) / 256, 256, 0, stream>>>(F(IN_XYZ), pts3, ptsA);
    gemm(stream, ptsA, 32, pc1, F(STN_C1 + 1), BN4(STN_BN1), BIGB, 64,  BN, 64,  GF_BN | GF_RELU, 0);
    gemm(stream, BIGB, 64, pc2, F(STN_C2 + 1), BN4(STN_BN2), BIGA, 128, BN, 128, GF_BN | GF_RELU, 0);
    hipMemsetAsync(hmax, 0, (size_t)16 * 1024 * sizeof(float), stream);
    gemm(stream, BIGA, 128, pc3, F(STN_C3 + 1), BN4(STN_BN3), hmax, 1024, BN, 1024,
         GF_BN | GF_RELU | GF_MAXRED, NPTS);
    k_cvt_bf16<<<(16 * 1024 + 255) / 256, 256, 0, stream>>>(hmax, hmaxbf, 16 * 1024);
    gemm(stream, hmaxbf, 1024, pf1, F(STN_F1 + 1), BN4(STN_BN4), f1o, 512, 16, 512, GF_BN | GF_RELU, 0);
    gemm(stream, f1o,    512,  pf2, F(STN_F2 + 1), BN4(STN_BN5), f2o, 256, 16, 256, GF_BN | GF_RELU, 0);
    gemm(stream, f2o,    256,  pf3, F(STN_F3 + 1), nullptr, nullptr, nullptr, nullptr,
         f3o, 32, 16, 9, GF_F32OUT, 0);
    k_make_trans<<<1, 64, 0, stream>>>(f3o, trans, (float*)d_out + (size_t)BN * 7);
    k_apply_trans<<<(BN + 255) / 256, 256, 0, stream>>>(pts3, trans, l0p);

    // --- Set abstraction ---
    static const int sa1o[3] = {32, 32, 64};
    static const int sa2o[3] = {64, 64, 128};
    static const int sa3o[3] = {128, 128, 256};
    static const int sa4o[3] = {256, 256, 512};
    run_sa(stream, d_in, pts3,   NPTS, l0p,  3,   1024, 0.1f, 32, SA1, sa1o, 3, psa1,
           l1_xyz, l1_p, fpsidx, ballidx, dist, BIGA, BIGB);
    run_sa(stream, d_in, l1_xyz, 1024, l1_p, 64,  256,  0.2f, 32, SA2, sa2o, 3, psa2,
           l2_xyz, l2_p, fpsidx, ballidx, dist, BIGA, BIGB);
    run_sa(stream, d_in, l2_xyz, 256,  l2_p, 128, 64,   0.4f, 32, SA3, sa3o, 3, psa3,
           l3_xyz, l3_p, fpsidx, ballidx, dist, BIGA, BIGB);
    run_sa(stream, d_in, l3_xyz, 64,   l3_p, 256, 16,   0.8f, 32, SA4, sa4o, 3, psa4,
           l4_xyz, l4_p, fpsidx, ballidx, dist, BIGA, BIGB);

    // --- Feature propagation ---
    static const int fp4o[2] = {256, 256};
    static const int fp3o[2] = {256, 256};
    static const int fp2o[2] = {256, 128};
    static const int fp1o[3] = {128, 128, 128};
    run_fp(stream, d_in, l3_xyz, 64,   l4_xyz, 16,   l3_p,    256, l4_p,  512, FP4, fp4o, 2, pfp4, BIGA, BIGB, l3_p2);
    run_fp(stream, d_in, l2_xyz, 256,  l3_xyz, 64,   l2_p,    128, l3_p2, 256, FP3, fp3o, 2, pfp3, BIGA, BIGB, l2_p2);
    run_fp(stream, d_in, l1_xyz, 1024, l2_xyz, 256,  l1_p,    64,  l2_p2, 256, FP2, fp2o, 2, pfp2, BIGA, BIGB, l1_p2);
    run_fp(stream, d_in, pts3,   NPTS, l1_xyz, 1024, nullptr, 0,   l1_p2, 128, FP1, fp1o, 3, pfp1, BIGA, BIGB, BIGB);

    // --- Head ---
    gemm(stream, BIGB, 128, ph1, F(HEAD1_B), F(HBN_G), F(HBN_B), F(HBN_M), F(HBN_V),
         BIGA, 128, BN, 128, GF_BN | GF_RELU, 0);
    gemm(stream, BIGA, 128, ph2, F(HEAD2_B), nullptr, nullptr, nullptr, nullptr,
         logits, 32, BN, 7, GF_F32OUT, 0);
    k_logsoftmax<<<(BN + 255) / 256, 256, 0, stream>>>(logits, 32, 7, BN, (float*)d_out);

    #undef F
    #undef BN4
}
